// Encoder_38259568672815
// MI455X (gfx1250) — compile-verified, matrix operand order checked by reading, not run on hardware
//
#include <hip/hip_runtime.h>
#include <math.h>

// ---------------------------------------------------------------------------
// ViT encoder for gfx1250 (MI455X).
//  * All matmuls through v_wmma_f32_16x16x32_bf16 (f32 accumulate).
//  * Weights converted once per launch to bf16 [n][k]; activations carry a
//    bf16 shadow copy, so all LDS tile staging is a pure byte copy ->
//    GLOBAL_LOAD_ASYNC_TO_LDS_B64 (ASYNCcnt) when the toolchain exposes it.
//  * Double-buffered LDS tiles: async copy of tile k+1 overlaps WMMA on tile k.
// ---------------------------------------------------------------------------

typedef __attribute__((ext_vector_type(16))) __bf16 bf16x16;
typedef __attribute__((ext_vector_type(8)))  float  f32x8;

#define BDIM   16
#define CCH    3
#define IMG    384
#define PSZ    16
#define DM     768
#define NHEAD  12
#define DKH    64
#define NLAYER 6
#define FFD    3072
#define STOK   576
#define MTOT   (BDIM*STOK)      // 9216

// ---------------- CDNA5 async global->LDS copy (guarded) -------------------

#if defined(__gfx1250__) && __has_builtin(__builtin_amdgcn_global_load_async_to_lds_b64) && __has_builtin(__builtin_amdgcn_s_wait_asynccnt)
#define ASYNC_COPY 1
#else
#define ASYNC_COPY 0
#endif

#if ASYNC_COPY
typedef int v2i __attribute__((ext_vector_type(2)));
typedef __attribute__((address_space(1))) v2i* as1_v2i;
typedef __attribute__((address_space(3))) v2i* as3_v2i;
#endif

__device__ __forceinline__ void copy8_g2l(const __bf16* gsrc, __bf16* ldst) {
#if ASYNC_COPY
    __builtin_amdgcn_global_load_async_to_lds_b64(
        (as1_v2i)(void*)gsrc, (as3_v2i)(void*)ldst, 0, 0);
#else
    *(int2*)ldst = *(const int2*)gsrc;          // fallback through VGPRs
#endif
}
__device__ __forceinline__ void copy_wait() {
#if ASYNC_COPY
    __builtin_amdgcn_s_wait_asynccnt(0);        // ASYNCcnt == 0
#endif
    // LDS visibility for the fallback path is covered by __syncthreads()
}

// ---------------- WMMA helpers ---------------------------------------------

__device__ __forceinline__ f32x8 wmma_bf16(bf16x16 a, bf16x16 b, f32x8 c) {
    return __builtin_amdgcn_wmma_f32_16x16x32_bf16(
        false, a, false, b, (short)0, c, false, false);
}

// 16x32 bf16 fragment from LDS (ISA 7.12.2): lanes 0-15 row M=lane hold
// K=0..7 / 16..23; lanes 16-31 hold K=8..15 / 24..31.  ld*2 must be %16==0.
__device__ __forceinline__ bf16x16 load_frag(const __bf16* base, int ld, int lane) {
    int half = lane >> 4;
    int r    = lane & 15;
    const __bf16* p = base + (size_t)r * ld + half * 8;
    union { bf16x16 v; int4 q[2]; } u;
    u.q[0] = *(const int4*)(p);
    u.q[1] = *(const int4*)(p + 16);
    return u.v;
}

// ---------------- GEMM -----------------------------------------------------
// C = act(A[M,K](bf16) @ Wt[N,K](bf16)^T + bias) (+residual), outputs f32
// and/or bf16.  128x128x32 tiles, 8 waves (2x4), double-buffered LDS.

#define BM 128
#define BN 128
#define BK 32
#define LDT 40      // 32 + 8 pad ; 80B row keeps 16B alignment

#define ACT_NONE 0
#define ACT_TANH 1
#define ACT_GELU 2

__global__ __launch_bounds__(256)
void gemm_bf16_kernel(const __bf16* __restrict__ A,   // [M][K]
                      const __bf16* __restrict__ Wt,  // [N][K]
                      const float*  __restrict__ bias,
                      const float*  __restrict__ residual,
                      float*  __restrict__ Cf,        // optional f32 out
                      __bf16* __restrict__ Cb,        // optional bf16 out
                      int M, int N, int K, int act)
{
    __shared__ __bf16 As[2][BM * LDT];
    __shared__ __bf16 Bs[2][BN * LDT];

    const int tid  = threadIdx.x;
    const int lane = tid & 31;
    const int wave = tid >> 5;
    const int wm   = wave & 1;       // 2 waves over M  -> 64 rows each
    const int wn   = wave >> 1;      // 4 waves over N  -> 32 cols each
    const int m0   = blockIdx.y * BM;
    const int n0   = blockIdx.x * BN;

    f32x8 acc[4][2];
    for (int mt = 0; mt < 4; ++mt)
        for (int nt = 0; nt < 2; ++nt)
            for (int r = 0; r < 8; ++r) acc[mt][nt][r] = 0.0f;

    // stage one 128x32 bf16 tile pair into buffer s (pure 8B copies)
    auto stage = [&](int s, int kb) {
        for (int i = tid; i < BM * 8; i += 256) {        // 8 x 8B chunks per row
            int row = i >> 3, ch = (i & 7) * 4;
            const __bf16* src = A + (size_t)(m0 + row) * K + kb + ch;
            copy8_g2l(src, &As[s][row * LDT + ch]);
            if (kb + BK < K)                              // warm L2 one tile ahead
                __builtin_prefetch((const void*)(src + BK), 0, 3);
        }
        for (int i = tid; i < BN * 8; i += 256) {
            int row = i >> 3, ch = (i & 7) * 4;
            const __bf16* src = Wt + (size_t)(n0 + row) * K + kb + ch;
            copy8_g2l(src, &Bs[s][row * LDT + ch]);
            if (kb + BK < K)
                __builtin_prefetch((const void*)(src + BK), 0, 3);
        }
    };

    stage(0, 0);
    int cur = 0;
    for (int kb = 0; kb < K; kb += BK) {
        copy_wait();
        __syncthreads();                       // buf[cur] ready for everyone
        if (kb + BK < K) stage(cur ^ 1, kb + BK);

        bf16x16 af[4], bfr[2];
        for (int mt = 0; mt < 4; ++mt)
            af[mt] = load_frag(&As[cur][(wm * 64 + mt * 16) * LDT], LDT, lane);
        for (int nt = 0; nt < 2; ++nt)
            bfr[nt] = load_frag(&Bs[cur][(wn * 32 + nt * 16) * LDT], LDT, lane);
        for (int mt = 0; mt < 4; ++mt)
            for (int nt = 0; nt < 2; ++nt)
                acc[mt][nt] = wmma_bf16(af[mt], bfr[nt], acc[mt][nt]);
        cur ^= 1;
    }

    // epilogue: vgpr r -> row r + 8*half, col = lane&15
    const int half = lane >> 4;
    for (int mt = 0; mt < 4; ++mt) {
        for (int nt = 0; nt < 2; ++nt) {
            int gmb = m0 + wm * 64 + mt * 16 + half * 8;
            int gn  = n0 + wn * 32 + nt * 16 + (lane & 15);
            float bv = bias ? bias[gn] : 0.0f;
            for (int r = 0; r < 8; ++r) {
                float v = acc[mt][nt][r] + bv;
                if (act == ACT_TANH)      v = tanhf(v);
                else if (act == ACT_GELU) v = 0.5f * v * (1.0f + erff(v * 0.70710678118654752f));
                size_t idx = (size_t)(gmb + r) * N + gn;
                if (residual) v += residual[idx];
                if (Cf) Cf[idx] = v;
                if (Cb) Cb[idx] = (__bf16)v;
            }
        }
    }
}

// ---------------- Flash attention ------------------------------------------
// one WG per (b, head, 64-query block); Q/K/V are bf16; 1/sqrt(dk) folded
// into the softmax read so Q/K staging stays a pure async copy.

__global__ __launch_bounds__(256)
void attn_kernel(const __bf16* __restrict__ Q, const __bf16* __restrict__ Kg,
                 const __bf16* __restrict__ Vg, __bf16* __restrict__ H)
{
    const int qb = blockIdx.x;    // 0..8
    const int h  = blockIdx.y;    // 0..11
    const int b  = blockIdx.z;    // 0..15

    __shared__ __bf16 Qs[64 * 72];    // [q][d]
    __shared__ __bf16 Ks[64 * 72];    // [key][d]
    __shared__ __bf16 Vt[64 * 72];    // [d][key]
    __shared__ __bf16 Pb[64 * 72];    // [q][key]
    __shared__ float  Sb[64 * 68];
    __shared__ float  rm[64], rl[64], al[64];

    const int tid  = threadIdx.x;
    const int lane = tid & 31;
    const int wave = tid >> 5;
    const int wm   = wave & 3;        // 4 waves over queries
    const int wn   = wave >> 2;       // 2 waves over keys / dims
    const int half = lane >> 4;
    const float scale = 0.125f;       // 1/sqrt(64)

    const size_t baseQ = ((size_t)b * STOK + (size_t)qb * 64) * DM + (size_t)h * DKH;

    for (int i = tid; i < 64 * 16; i += 256) {           // 16 x 8B chunks per row
        int row = i >> 4, ch = (i & 15) * 4;
        copy8_g2l(Q + baseQ + (size_t)row * DM + ch, &Qs[row * 72 + ch]);
    }
    if (tid < 64) { rm[tid] = -1e30f; rl[tid] = 0.0f; }

    f32x8 o[2];
    for (int nt = 0; nt < 2; ++nt)
        for (int r = 0; r < 8; ++r) o[nt][r] = 0.0f;

    copy_wait();
    __syncthreads();

    for (int kt = 0; kt < 9; ++kt) {
        const size_t baseK = ((size_t)b * STOK + (size_t)kt * 64) * DM + (size_t)h * DKH;
        for (int i = tid; i < 64 * 16; i += 256) {
            int row = i >> 4, ch = (i & 15) * 4;
            copy8_g2l(Kg + baseK + (size_t)row * DM + ch, &Ks[row * 72 + ch]);
            // V needs a transpose -> manual staged copy
            int2 w = *(const int2*)(Vg + baseK + (size_t)row * DM + ch);
            const __bf16* we = (const __bf16*)&w;
            Vt[(ch + 0) * 72 + row] = we[0];
            Vt[(ch + 1) * 72 + row] = we[1];
            Vt[(ch + 2) * 72 + row] = we[2];
            Vt[(ch + 3) * 72 + row] = we[3];
        }
        copy_wait();
        __syncthreads();

        // S = Q @ K^T
        f32x8 s[2];
        for (int nt = 0; nt < 2; ++nt)
            for (int r = 0; r < 8; ++r) s[nt][r] = 0.0f;
        for (int kk = 0; kk < DKH; kk += 32) {
            bf16x16 aq = load_frag(&Qs[(wm * 16) * 72 + kk], 72, lane);
            for (int nt = 0; nt < 2; ++nt) {
                bf16x16 bk = load_frag(&Ks[(wn * 32 + nt * 16) * 72 + kk], 72, lane);
                s[nt] = wmma_bf16(aq, bk, s[nt]);
            }
        }
        for (int nt = 0; nt < 2; ++nt)
            for (int r = 0; r < 8; ++r)
                Sb[(wm * 16 + r + half * 8) * 68 + wn * 32 + nt * 16 + (lane & 15)] = s[nt][r];
        __syncthreads();

        // online softmax: 4 threads per query row, scale folded in here
        {
            int row = tid >> 2;
            int l4  = tid & 3;
            const float* srow = &Sb[row * 68 + l4 * 16];
            float mx = -1e30f;
            for (int c = 0; c < 16; ++c) mx = fmaxf(mx, srow[c] * scale);
            mx = fmaxf(mx, __shfl_xor(mx, 1, 4));
            mx = fmaxf(mx, __shfl_xor(mx, 2, 4));
            float rmold = rm[row];
            float nm = fmaxf(rmold, mx);
            __bf16* prow = &Pb[row * 72 + l4 * 16];
            float sum = 0.0f;
            for (int c = 0; c < 16; ++c) {
                float p = __expf(srow[c] * scale - nm);
                sum += p;
                prow[c] = (__bf16)p;
            }
            sum += __shfl_xor(sum, 1, 4);
            sum += __shfl_xor(sum, 2, 4);
            float a = __expf(rmold - nm);
            if (l4 == 0) { rm[row] = nm; rl[row] = rl[row] * a + sum; al[row] = a; }
        }
        __syncthreads();

        // rescale O, then O += P @ V
        for (int r = 0; r < 8; ++r) {
            float a = al[wm * 16 + r + half * 8];
            o[0][r] *= a;
            o[1][r] *= a;
        }
        for (int kk = 0; kk < 64; kk += 32) {
            bf16x16 ap = load_frag(&Pb[(wm * 16) * 72 + kk], 72, lane);
            for (int nt = 0; nt < 2; ++nt) {
                bf16x16 bv = load_frag(&Vt[(wn * 32 + nt * 16) * 72 + kk], 72, lane);
                o[nt] = wmma_bf16(ap, bv, o[nt]);
            }
        }
        __syncthreads();
    }

    for (int nt = 0; nt < 2; ++nt) {
        for (int r = 0; r < 8; ++r) {
            int row = wm * 16 + r + half * 8;
            float v = o[nt][r] / rl[row];
            int st  = qb * 64 + row;
            int dd  = h * DKH + wn * 32 + nt * 16 + (lane & 15);
            H[((size_t)b * STOK + st) * DM + dd] = (__bf16)v;
        }
    }
}

// ---------------- LayerNorm (one block per token) ---------------------------

__global__ __launch_bounds__(256)
void ln_kernel(const float* __restrict__ X, float* __restrict__ Yf,
               __bf16* __restrict__ Yb, const float* __restrict__ sc,
               const float* __restrict__ bi, float eps)
{
    const int t   = blockIdx.x;
    const int tid = threadIdx.x;
    const float* x = X + (size_t)t * DM;
    float l[3], sum = 0.0f, sq = 0.0f;
    for (int j = 0; j < 3; ++j) {
        float v = x[tid + j * 256];
        l[j] = v; sum += v; sq += v * v;
    }
    for (int off = 16; off > 0; off >>= 1) {
        sum += __shfl_xor(sum, off, 32);
        sq  += __shfl_xor(sq,  off, 32);
    }
    __shared__ float ssum[8], ssq[8];
    if ((tid & 31) == 0) { ssum[tid >> 5] = sum; ssq[tid >> 5] = sq; }
    __syncthreads();
    float ts = 0.0f, tq = 0.0f;
    for (int i = 0; i < 8; ++i) { ts += ssum[i]; tq += ssq[i]; }
    float mean = ts * (1.0f / DM);
    float var  = tq * (1.0f / DM) - mean * mean;
    float inv  = rsqrtf(var + eps);
    for (int j = 0; j < 3; ++j) {
        int c = tid + j * 256;
        float y = (l[j] - mean) * inv * sc[c] + bi[c];
        if (Yf) Yf[(size_t)t * DM + c] = y;
        if (Yb) Yb[(size_t)t * DM + c] = (__bf16)y;
    }
}

// ---------------- weight / input prep ---------------------------------------

__global__ __launch_bounds__(256)
void convert_kernel(const float* __restrict__ W, __bf16* __restrict__ Wt, size_t n)
{
    size_t i = (size_t)blockIdx.x * 256 + threadIdx.x;
    if (i < n) Wt[i] = (__bf16)W[i];
}

// W[k][n] (f32) -> Wt[n][k] (bf16)
__global__ __launch_bounds__(256)
void convert_wt_kernel(const float* __restrict__ W, __bf16* __restrict__ Wt, int K, int N)
{
    size_t i = (size_t)blockIdx.x * 256 + threadIdx.x;
    if (i >= (size_t)K * N) return;
    int n = (int)(i / K);
    int k = (int)(i % K);
    Wt[i] = (__bf16)W[(size_t)k * N + n];
}

__global__ __launch_bounds__(256)
void im2col_kernel(const float* __restrict__ x, __bf16* __restrict__ Pm)
{
    size_t i = (size_t)blockIdx.x * 256 + threadIdx.x;
    if (i >= (size_t)MTOT * DM) return;
    int k = (int)(i % DM);
    int m = (int)(i / DM);
    int b = m / STOK, pos = m % STOK;
    int py = pos / 24, px = pos % 24;
    int c  = k >> 8;
    int ph = (k >> 4) & 15;
    int pw = k & 15;
    Pm[i] = (__bf16)x[(((size_t)b * CCH + c) * IMG + py * PSZ + ph) * IMG + px * PSZ + pw];
}

// faithful to the reference's no-transpose reshape + sinusoidal PE
__global__ __launch_bounds__(256)
void permute_pe_kernel(const float* __restrict__ E, float* __restrict__ X,
                       __bf16* __restrict__ Xb)
{
    size_t i = (size_t)blockIdx.x * 256 + threadIdx.x;
    if (i >= (size_t)BDIM * STOK * DM) return;
    int f = (int)(i % (STOK * DM));
    int b = (int)(i / (STOK * DM));
    int s  = f / DM;
    int d  = f % DM;
    int ch  = f / STOK;
    int pos = f % STOK;
    float v = E[((size_t)b * STOK + pos) * DM + ch];
    int ii = d >> 1;
    float div = __expf((float)(2 * ii) * (-9.210340371976184f / (float)DM));
    float ang = (float)s * div;
    v += (d & 1) ? __cosf(ang) : __sinf(ang);
    X[i]  = v;
    Xb[i] = (__bf16)v;
}

// ---------------- host launcher --------------------------------------------

static inline void launch_gemm(const __bf16* A, const __bf16* Wt, const float* bias,
                               const float* res, float* Cf, __bf16* Cb,
                               int M, int N, int K, int act, hipStream_t stream)
{
    dim3 grid(N / BN, M / BM);
    gemm_bf16_kernel<<<grid, 256, 0, stream>>>(A, Wt, bias, res, Cf, Cb, M, N, K, act);
}

extern "C" void kernel_launch(void* const* d_in, const int* in_sizes, int n_in,
                              void* d_out, int out_size, void* d_ws, size_t ws_size,
                              hipStream_t stream)
{
    (void)in_sizes; (void)n_in; (void)out_size; (void)ws_size;

    const float* x      = (const float*)d_in[0];
    const float* conv_w = (const float*)d_in[1];
    const float* conv_b = (const float*)d_in[2];
    // d_in[3], d_in[4]: ln1 scale/bias — unused by the reference
    const float* wq   = (const float*)d_in[5];
    const float* wk   = (const float*)d_in[6];
    const float* wv   = (const float*)d_in[7];
    const float* wh   = (const float*)d_in[8];
    const float* wh_b = (const float*)d_in[9];
    const float* ln2s = (const float*)d_in[10];
    const float* ln2b = (const float*)d_in[11];
    const float* w1   = (const float*)d_in[12];
    const float* b1   = (const float*)d_in[13];
    const float* w2   = (const float*)d_in[14];
    const float* b2   = (const float*)d_in[15];
    const float* lnfs = (const float*)d_in[16];
    const float* lnfb = (const float*)d_in[17];
    float* out = (float*)d_out;

    // ---- workspace layout (bytes) ----
    const size_t SXD   = (size_t)MTOT * DM;                 // 7,077,888 elems
    const size_t SFF   = (size_t)MTOT * FFD;                // 28,311,552 elems
    char* p = (char*)d_ws;
    float*  X    = (float*)p;              p += SXD * 4;
    __bf16* Xb   = (__bf16*)p;             p += SXD * 2;
    __bf16* Qb   = (__bf16*)p;             p += SXD * 2;
    __bf16* Kb   = (__bf16*)p;             p += SXD * 2;
    __bf16* Vb   = (__bf16*)p;             p += SXD * 2;
    __bf16* Hb   = (__bf16*)p;             p += SXD * 2;
    __bf16* Tb   = (__bf16*)p;             p += SXD * 2;
    __bf16* Fb   = (__bf16*)p;             p += SFF * 2;
    __bf16* WcT  = (__bf16*)p;             p += (size_t)DM * DM * 2;
    __bf16* WtL  = (__bf16*)p;             // 6 layers x 7,077,888 bf16 elems

    const size_t L_DD = (size_t)DM * DM;       // 589,824
    const size_t L_DF = (size_t)DM * FFD;      // 2,359,296
    const size_t LSTR = 4 * L_DD + 2 * L_DF;   // per-layer bf16 elems

    const int ELT_BLK = (int)((SXD + 255) / 256);
    const int DD_BLK  = (int)((L_DD + 255) / 256);
    const int DF_BLK  = (int)((L_DF + 255) / 256);

    // ---- weight prep: bf16, [n][k] ----
    // conv_w is (D, C*P*P) row-major == already [n][k]: straight convert
    convert_kernel<<<DD_BLK, 256, 0, stream>>>(conv_w, WcT, L_DD);
    for (int i = 0; i < NLAYER; ++i) {
        __bf16* wl = WtL + (size_t)i * LSTR;
        convert_wt_kernel<<<DD_BLK, 256, 0, stream>>>(wq + (size_t)i * L_DD, wl + 0 * L_DD, DM, DM);
        convert_wt_kernel<<<DD_BLK, 256, 0, stream>>>(wk + (size_t)i * L_DD, wl + 1 * L_DD, DM, DM);
        convert_wt_kernel<<<DD_BLK, 256, 0, stream>>>(wv + (size_t)i * L_DD, wl + 2 * L_DD, DM, DM);
        convert_wt_kernel<<<DD_BLK, 256, 0, stream>>>(wh + (size_t)i * L_DD, wl + 3 * L_DD, DM, DM);
        convert_wt_kernel<<<DF_BLK, 256, 0, stream>>>(w1 + (size_t)i * L_DF, wl + 4 * L_DD, DM, FFD);
        convert_wt_kernel<<<DF_BLK, 256, 0, stream>>>(w2 + (size_t)i * L_DF, wl + 4 * L_DD + L_DF, FFD, DM);
    }

    // ---- patch embedding ----
    __bf16* PmB = Qb;                 // im2col matrix (pre-layer scratch)
    float*  Ef  = (float*)Fb;         // conv output f32 (pre-layer scratch)
    im2col_kernel<<<ELT_BLK, 256, 0, stream>>>(x, PmB);
    launch_gemm(PmB, WcT, conv_b, nullptr, Ef, nullptr, MTOT, DM, DM, ACT_TANH, stream);
    permute_pe_kernel<<<ELT_BLK, 256, 0, stream>>>(Ef, X, Xb);

    // ---- encoder layers ----
    for (int i = 0; i < NLAYER; ++i) {
        __bf16* wl   = WtL + (size_t)i * LSTR;
        __bf16* wqT  = wl + 0 * L_DD;
        __bf16* wkT  = wl + 1 * L_DD;
        __bf16* wvT  = wl + 2 * L_DD;
        __bf16* whT  = wl + 3 * L_DD;
        __bf16* w1T  = wl + 4 * L_DD;
        __bf16* w2T  = wl + 4 * L_DD + L_DF;

        launch_gemm(Xb, wqT, nullptr, nullptr, nullptr, Qb, MTOT, DM, DM, ACT_NONE, stream);
        launch_gemm(Xb, wkT, nullptr, nullptr, nullptr, Kb, MTOT, DM, DM, ACT_NONE, stream);
        launch_gemm(Xb, wvT, nullptr, nullptr, nullptr, Vb, MTOT, DM, DM, ACT_NONE, stream);

        attn_kernel<<<dim3(STOK / 64, NHEAD, BDIM), 256, 0, stream>>>(Qb, Kb, Vb, Hb);

        // x1 = H @ wh + wh_b + x  (f32, in place)
        launch_gemm(Hb, whT, wh_b + (size_t)i * DM, X, X, nullptr, MTOT, DM, DM, ACT_NONE, stream);

        // xn = LN2(x1) -> bf16 only (feeds FF1)
        ln_kernel<<<MTOT, 256, 0, stream>>>(X, nullptr, Tb, ln2s + (size_t)i * DM,
                                            ln2b + (size_t)i * DM, 1e-6f);

        // ff = gelu(xn @ w1 + b1) -> bf16 only
        launch_gemm(Tb, w1T, b1 + (size_t)i * FFD, nullptr, nullptr, Fb,
                    MTOT, FFD, DM, ACT_GELU, stream);
        // x = x1 + ff @ w2 + b2 -> f32 (residual/LN) + bf16 (next layer QKV)
        launch_gemm(Fb, w2T, b2 + (size_t)i * DM, X, X, Xb,
                    MTOT, DM, FFD, ACT_NONE, stream);
    }

    ln_kernel<<<MTOT, 256, 0, stream>>>(X, out, nullptr, lnfs, lnfb, 1e-12f);
}
